// Qwen2MoeDecoderLayer_49091476193839
// MI455X (gfx1250) — matrix operand order repaired and checked by reading, not compiled, not run on hardware
//
#include <hip/hip_runtime.h>

// ---------------- types & constants ----------------
typedef __attribute__((ext_vector_type(16))) __bf16 v16bf;
typedef __attribute__((ext_vector_type(8)))  float  v8f;
typedef __attribute__((ext_vector_type(4)))  int    v4i;

#define TT   2048   // tokens
#define HH   2048   // hidden
#define NHEAD 16
#define HD   128    // head dim
#define NEXP 16
#define TOPK 4
#define IMOE 1408
#define ISH  5632

#define AS1 __attribute__((address_space(1)))
#define AS3 __attribute__((address_space(3)))

// gfx1250 async global->LDS staging path (ASYNCcnt-tracked, bypasses VGPRs).
#if defined(__HIP_DEVICE_COMPILE__) && defined(__gfx1250__) && \
    __has_builtin(__builtin_amdgcn_global_load_async_to_lds_b128)
#define USE_ASYNC_LDS 1
#endif

#ifdef USE_ASYNC_LDS
// 16B global -> LDS async transfer (GLOBAL_LOAD_ASYNC_TO_LDS_B128).
__device__ inline void async_cp16(const __bf16* g, __bf16* l) {
  __builtin_amdgcn_global_load_async_to_lds_b128(
      (AS1 v4i*)g, (AS3 v4i*)l, 0, 0);
}
#endif

__device__ inline void async_wait0() {
#ifdef USE_ASYNC_LDS
#if __has_builtin(__builtin_amdgcn_s_wait_asynccnt)
  __builtin_amdgcn_s_wait_asynccnt(0);
#else
  asm volatile("s_wait_asynccnt 0" ::: "memory");
#endif
#endif
}

// Load a 16-half WMMA fragment as two contiguous b128 LDS loads.
__device__ inline v16bf ld_frag(const __bf16* p0, const __bf16* p1) {
  union { v16bf v; uint4 u[2]; } t;
  t.u[0] = *(const uint4*)p0;
  t.u[1] = *(const uint4*)p1;
  return t.v;
}

// ---------------- fused add + RMSNorm (writes res f32 + normed bf16) --------
__global__ __launch_bounds__(256) void add_rmsnorm_k(
    const float* __restrict__ a, const float* __restrict__ b,
    const float* __restrict__ w, float* __restrict__ res_out,
    __bf16* __restrict__ hb) {
  const int t = blockIdx.x;
  const int tid = threadIdx.x;
  __shared__ float red[256];
  float vals[8];
  float ss = 0.f;
  #pragma unroll
  for (int i = 0; i < 8; ++i) {
    int c = tid + i * 256;
    float v = a[(size_t)t * HH + c] + b[(size_t)t * HH + c];
    vals[i] = v;
    ss += v * v;
  }
  red[tid] = ss;
  __syncthreads();
  for (int st = 128; st > 0; st >>= 1) {
    if (tid < st) red[tid] += red[tid + st];
    __syncthreads();
  }
  float rstd = rsqrtf(red[0] / (float)HH + 1e-6f);
  #pragma unroll
  for (int i = 0; i < 8; ++i) {
    int c = tid + i * 256;
    res_out[(size_t)t * HH + c] = vals[i];
    hb[(size_t)t * HH + c] = (__bf16)(vals[i] * rstd * w[c]);
  }
}

// ---------------- generic bf16 WMMA GEMM: C = A @ W^T (+bias) ----------------
// A: bf16 [rows, Kd] row-major (optionally gathered).  W: f32 [N, Kd] row-major,
// converted to bf16 while staging into LDS.  Double-buffered K-loop; A tile is
// staged with gfx1250 async global->LDS b128 transfers when available.
// GROUPED: blockIdx.z = expert group with per-group count/offset (+ optional
// gather list); output rows compact at rowBase+m.
template <bool GROUPED>
__global__ __launch_bounds__(256) void gemm_bf16_wmma(
    const __bf16* __restrict__ A, const float* __restrict__ W,
    const float* __restrict__ bias, float* __restrict__ Cf,
    __bf16* __restrict__ Cb, int M, int N, int Kd,
    const int* __restrict__ gather, const int* __restrict__ grp_cnt,
    const int* __restrict__ grp_off, long long wstride) {
  const int e = blockIdx.z;
  const int mcount = GROUPED ? grp_cnt[e] : M;
  const int rowBase = GROUPED ? grp_off[e] : 0;
  const int blockRow = blockIdx.y * 128;
  if (GROUPED && blockRow >= mcount) return;      // uniform per-block exit
  const float* Wg = GROUPED ? (W + (size_t)e * (size_t)wstride) : W;
  const int nBase = blockIdx.x * 64;
  __shared__ __bf16 As[2][128][40];               // [m][k], pitch 80B
  __shared__ __bf16 Bw[2][64][40];                // [n][k] (un-transposed W tile)
  const int tid = threadIdx.x;
  const int lane = tid & 31, wvid = tid >> 5;
  const int wm = wvid & 3, wn = wvid >> 2;        // 4x2 wave grid (8 waves)
  const int l16 = lane & 15, hi = lane >> 4;
  const bool full = !GROUPED || (blockRow + 128 <= mcount);

  auto stageA = [&](int buf, int k0) {
    if (full) {
#ifdef USE_ASYNC_LDS
      for (int s = tid; s < 128 * 4; s += 256) {  // 4 x b128 per 64B row
        int r = s >> 2, c8 = (s & 3) << 3;
        int gr = blockRow + r;
        size_t arow = GROUPED
            ? (size_t)(gather ? gather[rowBase + gr] : (rowBase + gr))
            : (size_t)gr;
        async_cp16(A + arow * Kd + k0 + c8, &As[buf][r][c8]);
      }
#else
      for (int s = tid; s < 128 * 8; s += 256) {
        int r = s >> 3, c4 = (s & 7) << 2;
        int gr = blockRow + r;
        size_t arow = GROUPED
            ? (size_t)(gather ? gather[rowBase + gr] : (rowBase + gr))
            : (size_t)gr;
        *(uint2*)&As[buf][r][c4] = *(const uint2*)(A + arow * Kd + k0 + c4);
      }
#endif
    } else {  // tail block of a group: guarded, zero-fill
      for (int s = tid; s < 128 * 8; s += 256) {
        int r = s >> 3, c4 = (s & 7) << 2;
        int gr = blockRow + r;
        uint2 val; val.x = 0u; val.y = 0u;
        if (gr < mcount) {
          int arow = gather ? gather[rowBase + gr] : (rowBase + gr);
          val = *(const uint2*)(A + (size_t)arow * Kd + k0 + c4);
        }
        *(uint2*)&As[buf][r][c4] = val;
      }
    }
  };
  auto stageB = [&](int buf, int k0) {
    for (int s = tid; s < 64 * 8; s += 256) {
      int n = s >> 3, c4 = (s & 7) << 2;
      const float* wp = Wg + (size_t)(nBase + n) * Kd + k0 + c4;
      float4 w4 = *(const float4*)wp;
      if (k0 + 32 < Kd) __builtin_prefetch((const void*)(wp + 32), 0, 1);
      union { __bf16 h[4]; uint2 u; } pk;
      pk.h[0] = (__bf16)w4.x; pk.h[1] = (__bf16)w4.y;
      pk.h[2] = (__bf16)w4.z; pk.h[3] = (__bf16)w4.w;
      *(uint2*)&Bw[buf][n][c4] = pk.u;
    }
  };

  v8f acc[2][2];
  for (int i = 0; i < 2; ++i)
    for (int j = 0; j < 2; ++j)
      #pragma unroll
      for (int r = 0; r < 8; ++r) acc[i][j][r] = 0.f;

  stageA(0, 0);
  stageB(0, 0);
  int buf = 0;
  for (int k0 = 0; k0 < Kd; k0 += 32) {
    async_wait0();          // this wave's async stores to As[buf] are in LDS
    __syncthreads();        // all waves' tiles for `buf` published
    if (k0 + 32 < Kd) {     // overlap next-tile staging with current WMMAs
      stageA(buf ^ 1, k0 + 32);
      stageB(buf ^ 1, k0 + 32);
    }
    v16bf af[2], bfb[2];
    #pragma unroll
    for (int i = 0; i < 2; ++i) {   // A frags: row = M (lanes 0-15), K split by hi
      int row = wm * 32 + i * 16 + l16;
      af[i] = ld_frag(&As[buf][row][hi * 8], &As[buf][row][16 + hi * 8]);
    }
    #pragma unroll
    for (int i = 0; i < 2; ++i) {   // B frags: col = N, K split by hi
      int col = wn * 32 + i * 16 + l16;
      bfb[i] = ld_frag(&Bw[buf][col][hi * 8], &Bw[buf][col][16 + hi * 8]);
    }
    #pragma unroll
    for (int mi = 0; mi < 2; ++mi) {
      #pragma unroll
      for (int ni = 0; ni < 2; ++ni) {
        acc[mi][ni] = __builtin_amdgcn_wmma_f32_16x16x32_bf16(
            false, af[mi], false, bfb[ni], (short)0, acc[mi][ni], false, false);
      }
    }
    buf ^= 1;
  }
  // epilogue: C layout lane&15 = N col, VGPR r -> M row r + hi*8
  #pragma unroll
  for (int mi = 0; mi < 2; ++mi) {
    #pragma unroll
    for (int ni = 0; ni < 2; ++ni) {
      #pragma unroll
      for (int r = 0; r < 8; ++r) {
        int row = blockRow + wm * 32 + mi * 16 + hi * 8 + r;
        if (row < mcount) {
          int col = nBase + wn * 32 + ni * 16 + l16;
          float v = acc[mi][ni][r];
          if (bias) v += bias[col];
          size_t o = (size_t)(rowBase + row) * (size_t)N + col;
          if (Cf) Cf[o] = v;
          if (Cb) Cb[o] = (__bf16)v;
        }
      }
    }
  }
}

// ---------------- rope + qkv split to per-head bf16 [h][t][d] ----------------
__global__ __launch_bounds__(256) void rope_split_k(
    const float* __restrict__ qkv, const int* __restrict__ pos,
    __bf16* __restrict__ Qo, __bf16* __restrict__ Ko, __bf16* __restrict__ Vo) {
  int idx = blockIdx.x * 256 + threadIdx.x;       // T * NH * 64
  int d = idx & 63;
  int h = (idx >> 6) & (NHEAD - 1);
  int t = idx >> 10;
  if (t >= TT) return;
  // inv = ROPE_BASE^(-2d/HD) = exp(-2d/HD * ln(1e6))
  float inv = __expf(-(float)(2 * d) * (13.815510557964274f / (float)HD));
  float ang = (float)pos[t] * inv;
  float cs = __cosf(ang), sn = __sinf(ang);
  const float* row = qkv + (size_t)t * 3 * HH;
  size_t ob = ((size_t)h * TT + t) * HD;
  float x1 = row[h * HD + d], x2 = row[h * HD + d + 64];
  Qo[ob + d]      = (__bf16)(x1 * cs - x2 * sn);
  Qo[ob + d + 64] = (__bf16)(x2 * cs + x1 * sn);
  x1 = row[HH + h * HD + d]; x2 = row[HH + h * HD + d + 64];
  Ko[ob + d]      = (__bf16)(x1 * cs - x2 * sn);
  Ko[ob + d + 64] = (__bf16)(x2 * cs + x1 * sn);
  Vo[ob + d]      = (__bf16)row[2 * HH + h * HD + d];
  Vo[ob + d + 64] = (__bf16)row[2 * HH + h * HD + d + 64];
}

// ---------------- WMMA flash attention (causal, one head per blockIdx.y) ----
__global__ __launch_bounds__(128) void flash_attn_k(
    const __bf16* __restrict__ Q, const __bf16* __restrict__ Kc,
    const __bf16* __restrict__ Vc, __bf16* __restrict__ O) {
  const int head = blockIdx.y;
  const int qBase = blockIdx.x * 64;
  const int tid = threadIdx.x;
  const int lane = tid & 31, w = tid >> 5;        // 4 waves, 16 q-rows each
  const int l16 = lane & 15, hi = lane >> 4;
  __shared__ __bf16 Qs[64][136];                  // [q][d]
  __shared__ __bf16 Ks[64][136];                  // [k][d]
  __shared__ __bf16 Vt[128][72];                  // [d][k] (transposed V)
  __shared__ __bf16 Ps[4][16][72];                // per-wave P tile [m][k]

  const __bf16* Qh = Q + ((size_t)head * TT + qBase) * HD;
  for (int s = tid; s < 64 * 32; s += 128) {
    int r = s >> 5, c4 = (s & 31) << 2;
    *(uint2*)&Qs[r][c4] = *(const uint2*)(Qh + (size_t)r * HD + c4);
  }
  __syncthreads();
  v16bf aQ[4];
  #pragma unroll
  for (int c = 0; c < 4; ++c)
    aQ[c] = ld_frag(&Qs[w * 16 + l16][c * 32 + hi * 8],
                    &Qs[w * 16 + l16][c * 32 + 16 + hi * 8]);

  float mrow[8], lrow[8];
  v8f oacc[8];
  #pragma unroll
  for (int r = 0; r < 8; ++r) { mrow[r] = -1e30f; lrow[r] = 0.f; }
  for (int c = 0; c < 8; ++c)
    #pragma unroll
    for (int r = 0; r < 8; ++r) oacc[c][r] = 0.f;

  const float scale = 0.08838834764831845f;       // 1/sqrt(128)
  const __bf16* Kh = Kc + (size_t)head * TT * HD;
  const __bf16* Vh = Vc + (size_t)head * TT * HD;

  for (int kb = 0; kb <= qBase; kb += 64) {
    __syncthreads();
    for (int s = tid; s < 64 * 32; s += 128) {
      int r = s >> 5, c4 = (s & 31) << 2;
      *(uint2*)&Ks[r][c4] = *(const uint2*)(Kh + (size_t)(kb + r) * HD + c4);
      uint2 vv = *(const uint2*)(Vh + (size_t)(kb + r) * HD + c4);
      const __bf16* ve = (const __bf16*)&vv;
      Vt[c4 + 0][r] = ve[0]; Vt[c4 + 1][r] = ve[1];
      Vt[c4 + 2][r] = ve[2]; Vt[c4 + 3][r] = ve[3];
    }
    __syncthreads();
    // S = Q @ K^T  (4 key sub-tiles x 4 K-chunks)
    v8f sfr[4];
    #pragma unroll
    for (int nt = 0; nt < 4; ++nt) {
      v8f sv;
      #pragma unroll
      for (int r = 0; r < 8; ++r) sv[r] = 0.f;
      #pragma unroll
      for (int c = 0; c < 4; ++c) {
        v16bf bk = ld_frag(&Ks[nt * 16 + l16][c * 32 + hi * 8],
                           &Ks[nt * 16 + l16][c * 32 + 16 + hi * 8]);
        sv = __builtin_amdgcn_wmma_f32_16x16x32_bf16(
            false, aQ[c], false, bk, (short)0, sv, false, false);
      }
      sfr[nt] = sv;
    }
    // scale + causal mask
    #pragma unroll
    for (int nt = 0; nt < 4; ++nt) {
      int key = kb + nt * 16 + l16;
      #pragma unroll
      for (int r = 0; r < 8; ++r) {
        int qr = qBase + w * 16 + hi * 8 + r;
        float v = sfr[nt][r] * scale;
        sfr[nt][r] = (key > qr) ? -1e30f : v;
      }
    }
    // online softmax: row stats live across 16 lanes sharing `hi`
    float corr[8];
    #pragma unroll
    for (int r = 0; r < 8; ++r) {
      float mx = fmaxf(fmaxf(sfr[0][r], sfr[1][r]), fmaxf(sfr[2][r], sfr[3][r]));
      #pragma unroll
      for (int off = 1; off < 16; off <<= 1) mx = fmaxf(mx, __shfl_xor(mx, off, 32));
      float mn = fmaxf(mrow[r], mx);
      float co = __expf(mrow[r] - mn);
      float ps = 0.f;
      #pragma unroll
      for (int nt = 0; nt < 4; ++nt) {
        float p = __expf(sfr[nt][r] - mn);
        sfr[nt][r] = p;
        ps += p;
      }
      #pragma unroll
      for (int off = 1; off < 16; off <<= 1) ps += __shfl_xor(ps, off, 32);
      lrow[r] = lrow[r] * co + ps;
      mrow[r] = mn;
      corr[r] = co;
    }
    for (int c = 0; c < 8; ++c)
      #pragma unroll
      for (int r = 0; r < 8; ++r) oacc[c][r] *= corr[r];
    // repack P (C-layout) -> A-layout through wave-private LDS tile
    #pragma unroll
    for (int nt = 0; nt < 4; ++nt)
      #pragma unroll
      for (int r = 0; r < 8; ++r)
        Ps[w][hi * 8 + r][nt * 16 + l16] = (__bf16)sfr[nt][r];
    asm volatile("s_wait_dscnt 0" ::: "memory");  // wave-local LDS RAW
    v16bf aP[2];
    #pragma unroll
    for (int c2 = 0; c2 < 2; ++c2)
      aP[c2] = ld_frag(&Ps[w][l16][c2 * 32 + hi * 8],
                       &Ps[w][l16][c2 * 32 + 16 + hi * 8]);
    // O += P @ V
    #pragma unroll
    for (int c = 0; c < 8; ++c) {
      #pragma unroll
      for (int c2 = 0; c2 < 2; ++c2) {
        v16bf bv = ld_frag(&Vt[c * 16 + l16][c2 * 32 + hi * 8],
                           &Vt[c * 16 + l16][c2 * 32 + 16 + hi * 8]);
        oacc[c] = __builtin_amdgcn_wmma_f32_16x16x32_bf16(
            false, aP[c2], false, bv, (short)0, oacc[c], false, false);
      }
    }
  }
  // write O as bf16 [t][h*HD+d] for the o-proj GEMM
  #pragma unroll
  for (int c = 0; c < 8; ++c) {
    #pragma unroll
    for (int r = 0; r < 8; ++r) {
      int q = qBase + w * 16 + hi * 8 + r;
      float v = oacc[c][r] / lrow[r];
      O[(size_t)q * HH + head * HD + c * 16 + l16] = (__bf16)v;
    }
  }
}

// ---------------- router: fp32 softmax over 16 experts, top-4 ----------------
__global__ __launch_bounds__(256) void router_k(
    const __bf16* __restrict__ h2b, const float* __restrict__ gate_w,
    const float* __restrict__ sgw, float* __restrict__ topw,
    int* __restrict__ topi, float* __restrict__ sgate) {
  const int t = blockIdx.x;
  const int tid = threadIdx.x;
  __shared__ float part[256];
  __shared__ float logits[NEXP];
  const __bf16* hrow = h2b + (size_t)t * HH;
  int e = tid >> 4, j = tid & 15;
  float acc = 0.f;
  for (int k = j; k < HH; k += 16) acc += (float)hrow[k] * gate_w[(size_t)e * HH + k];
  part[tid] = acc;
  __syncthreads();
  if (j == 0) {
    float s = 0.f;
    for (int i = 0; i < 16; ++i) s += part[e * 16 + i];
    logits[e] = s;
  }
  __syncthreads();
  float a2 = 0.f;
  for (int k = tid; k < HH; k += 256) a2 += (float)hrow[k] * sgw[k];
  part[tid] = a2;
  __syncthreads();
  for (int st = 128; st > 0; st >>= 1) {
    if (tid < st) part[tid] += part[tid + st];
    __syncthreads();
  }
  if (tid == 0) {
    sgate[t] = part[0];
    float mx = logits[0];
    for (int i = 1; i < NEXP; ++i) mx = fmaxf(mx, logits[i]);
    float p[NEXP]; float s = 0.f;
    for (int i = 0; i < NEXP; ++i) { p[i] = __expf(logits[i] - mx); s += p[i]; }
    float invs = 1.f / s;
    bool used[NEXP];
    for (int i = 0; i < NEXP; ++i) used[i] = false;
    for (int kk = 0; kk < TOPK; ++kk) {
      int best = 0; float bv = -1.f;
      for (int i = 0; i < NEXP; ++i) {
        float pv = p[i] * invs;
        if (!used[i] && pv > bv) { bv = pv; best = i; }
      }
      used[best] = true;
      topw[t * TOPK + kk] = bv;
      topi[t * TOPK + kk] = best;
    }
  }
}

// ---------------- deterministic token->expert dispatch ----------------------
__global__ void dispatch_k(const int* __restrict__ topi, int* __restrict__ counts,
                           int* __restrict__ offs, int* __restrict__ toklist,
                           int* __restrict__ slot_of) {
  __shared__ int cnt[NEXP];
  __shared__ int off[NEXP];
  int e = threadIdx.x;
  if (e < NEXP) {
    int c = 0;
    for (int t = 0; t < TT; ++t)
      for (int j = 0; j < TOPK; ++j)
        if (topi[t * TOPK + j] == e) ++c;
    cnt[e] = c;
  }
  __syncthreads();
  if (threadIdx.x == 0) {
    int s = 0;
    for (int i = 0; i < NEXP; ++i) { off[i] = s; s += cnt[i]; }
  }
  __syncthreads();
  if (e < NEXP) {
    int m = 0, base = off[e];
    for (int t = 0; t < TT; ++t)
      for (int j = 0; j < TOPK; ++j)
        if (topi[t * TOPK + j] == e) {
          toklist[base + m] = t;
          slot_of[t * TOPK + j] = base + m;
          ++m;
        }
    counts[e] = cnt[e];
    offs[e] = off[e];
  }
}

// ---------------- swiglu: act = silu(g) * u -----------------------------------
__global__ __launch_bounds__(256) void swiglu_k(const __bf16* __restrict__ gu,
                                                __bf16* __restrict__ act,
                                                int rows, int isz) {
  size_t idx = (size_t)blockIdx.x * 256 + threadIdx.x;
  if (idx >= (size_t)rows * isz) return;
  int r = (int)(idx / isz), i = (int)(idx % isz);
  float g = (float)gu[(size_t)r * 2 * isz + i];
  float u = (float)gu[(size_t)r * 2 * isz + isz + i];
  float sg = g / (1.f + __expf(-g));
  act[(size_t)r * isz + i] = (__bf16)(sg * u);
}

// ---------------- final: shared*sigmoid(gate) + top-k combine ----------------
__global__ __launch_bounds__(256) void final_combine_k(
    const float* __restrict__ shdown, const float* __restrict__ sgate,
    const float* __restrict__ topw, const int* __restrict__ slot_of,
    const float* __restrict__ ydown, float* __restrict__ out) {
  size_t idx = (size_t)blockIdx.x * 256 + threadIdx.x;
  int t = (int)(idx / HH), c = (int)(idx % HH);
  float g = 1.f / (1.f + __expf(-sgate[t]));
  float v = shdown[idx] * g;
  #pragma unroll
  for (int j = 0; j < TOPK; ++j)
    v += topw[t * TOPK + j] * ydown[(size_t)slot_of[t * TOPK + j] * HH + c];
  out[idx] = v;
}

// ---------------- host launcher ----------------------------------------------
extern "C" void kernel_launch(void* const* d_in, const int* in_sizes, int n_in,
                              void* d_out, int out_size, void* d_ws, size_t ws_size,
                              hipStream_t stream) {
  const int*   positions = (const int*)d_in[0];
  const float* hidden    = (const float*)d_in[1];
  const float* residual  = (const float*)d_in[2];
  const float* ln1_w     = (const float*)d_in[3];
  const float* ln2_w     = (const float*)d_in[4];
  const float* qkv_w     = (const float*)d_in[5];
  const float* qkv_b     = (const float*)d_in[6];
  const float* o_w       = (const float*)d_in[7];
  const float* gate_w    = (const float*)d_in[8];
  const float* sgate_w   = (const float*)d_in[9];
  const float* sh_gup_w  = (const float*)d_in[10];
  const float* sh_down_w = (const float*)d_in[11];
  const float* e_gup_w   = (const float*)d_in[12];
  const float* e_down_w  = (const float*)d_in[13];

  float* out1 = (float*)d_out;            // shared + sparse
  float* out2 = out1 + (size_t)TT * HH;   // res2

  size_t off = 0;
  auto alloc = [&](size_t bytes) -> void* {
    void* p = (char*)d_ws + off;
    off += (bytes + 255) & ~(size_t)255;
    return p;
  };
  float*  res    = (float*)alloc((size_t)TT * HH * 4);
  __bf16* hb     = (__bf16*)alloc((size_t)TT * HH * 2);
  float*  qkv    = (float*)alloc((size_t)TT * 3 * HH * 4);
  __bf16* qb     = (__bf16*)alloc((size_t)TT * HH * 2);
  __bf16* kb     = (__bf16*)alloc((size_t)TT * HH * 2);
  __bf16* vb     = (__bf16*)alloc((size_t)TT * HH * 2);
  __bf16* attnb  = (__bf16*)alloc((size_t)TT * HH * 2);
  float*  oout   = (float*)alloc((size_t)TT * HH * 4);
  __bf16* h2b    = (__bf16*)alloc((size_t)TT * HH * 2);
  float*  topw   = (float*)alloc((size_t)TT * TOPK * 4);
  int*    topi   = (int*)alloc((size_t)TT * TOPK * 4);
  float*  sgate  = (float*)alloc((size_t)TT * 4);
  int*    counts = (int*)alloc(NEXP * 4);
  int*    offs   = (int*)alloc(NEXP * 4);
  int*    toklst = (int*)alloc((size_t)TT * TOPK * 4);
  int*    slotof = (int*)alloc((size_t)TT * TOPK * 4);
  __bf16* shgup  = (__bf16*)alloc((size_t)TT * 2 * ISH * 2);
  __bf16* shact  = (__bf16*)alloc((size_t)TT * ISH * 2);
  float*  shdown = (float*)alloc((size_t)TT * HH * 4);
  __bf16* egup   = (__bf16*)alloc((size_t)TT * TOPK * 2 * IMOE * 2);
  __bf16* eact   = (__bf16*)alloc((size_t)TT * TOPK * IMOE * 2);
  float*  ydown  = (float*)alloc((size_t)TT * TOPK * HH * 4);
  (void)ws_size; (void)n_in; (void)in_sizes; (void)out_size;

  // 1. res = hidden + residual ; h = rmsnorm(res)
  add_rmsnorm_k<<<TT, 256, 0, stream>>>(hidden, residual, ln1_w, res, hb);
  // 2. qkv = h @ qkv_w^T + b  (M=2048, N=6144, K=2048)
  gemm_bf16_wmma<false><<<dim3(3 * HH / 64, TT / 128, 1), 256, 0, stream>>>(
      hb, qkv_w, qkv_b, qkv, nullptr, TT, 3 * HH, HH, nullptr, nullptr, nullptr, 0);
  // 3. rope + split to per-head bf16
  rope_split_k<<<(TT * NHEAD * 64) / 256, 256, 0, stream>>>(qkv, positions, qb, kb, vb);
  // 4. causal flash attention (WMMA)
  flash_attn_k<<<dim3(TT / 64, NHEAD), 128, 0, stream>>>(qb, kb, vb, attnb);
  // 5. o-proj
  gemm_bf16_wmma<false><<<dim3(HH / 64, TT / 128, 1), 256, 0, stream>>>(
      attnb, o_w, nullptr, oout, nullptr, TT, HH, HH, nullptr, nullptr, nullptr, 0);
  // 6. res2 = attn + res ; h2 = rmsnorm(res2)  (res2 -> output slot 2)
  add_rmsnorm_k<<<TT, 256, 0, stream>>>(oout, res, ln2_w, out2, h2b);
  // 7. router (fp32 softmax, top-4, no renorm) + shared-gate logit
  router_k<<<TT, 256, 0, stream>>>(h2b, gate_w, sgate_w, topw, topi, sgate);
  // 8. deterministic dispatch lists
  dispatch_k<<<1, 32, 0, stream>>>(topi, counts, offs, toklst, slotof);
  // 9. shared expert: gup -> swiglu -> down
  gemm_bf16_wmma<false><<<dim3(2 * ISH / 64, TT / 128, 1), 256, 0, stream>>>(
      h2b, sh_gup_w, nullptr, nullptr, shgup, TT, 2 * ISH, HH, nullptr, nullptr, nullptr, 0);
  swiglu_k<<<(unsigned)(((size_t)TT * ISH + 255) / 256), 256, 0, stream>>>(shgup, shact, TT, ISH);
  gemm_bf16_wmma<false><<<dim3(HH / 64, TT / 128, 1), 256, 0, stream>>>(
      shact, sh_down_w, nullptr, shdown, nullptr, TT, HH, ISH, nullptr, nullptr, nullptr, 0);
  // 10. sparse experts: grouped gathered gup -> swiglu -> grouped down
  gemm_bf16_wmma<true><<<dim3(2 * IMOE / 64, TT / 128, NEXP), 256, 0, stream>>>(
      h2b, e_gup_w, nullptr, nullptr, egup, TT, 2 * IMOE, HH,
      toklst, counts, offs, (long long)2 * IMOE * HH);
  swiglu_k<<<(unsigned)(((size_t)TT * TOPK * IMOE + 255) / 256), 256, 0, stream>>>(
      egup, eact, TT * TOPK, IMOE);
  gemm_bf16_wmma<true><<<dim3(HH / 64, TT / 128, NEXP), 256, 0, stream>>>(
      eact, e_down_w, nullptr, ydown, nullptr, TT, HH, IMOE,
      nullptr, counts, offs, (long long)HH * IMOE);
  // 11. out = shared * sigmoid(gate) + top-k combine
  final_combine_k<<<(TT * HH) / 256, 256, 0, stream>>>(
      shdown, sgate, topw, slotof, ydown, out1);
}
